// TextRNN2_32830730011275
// MI455X (gfx1250) — matrix operand order repaired and checked by reading, not compiled, not run on hardware
//
#include <hip/hip_runtime.h>

// Problem constants (from reference)
#define BB 64      // batch
#define SS 256     // sequence length
#define EE 512     // embed / layer-1 hidden
#define HH2 1024   // layer-2 hidden
#define LL 10      // classes

typedef __attribute__((ext_vector_type(16))) __bf16 v16bf;
typedef __attribute__((ext_vector_type(8)))  float  v8f;
typedef unsigned short ushort_t;
typedef unsigned int   uint_t;

__device__ __forceinline__ ushort_t f2bf(float x) {
  uint_t u = __float_as_uint(x);
  u += 0x7FFFu + ((u >> 16) & 1u);          // round-to-nearest-even
  return (ushort_t)(u >> 16);
}

__device__ __forceinline__ float sigm(float x) { return 1.0f / (1.0f + __expf(-x)); }

union FragU { v16bf v; uint_t u[8]; uint4 q[2]; };

// A-fragment (16x32 bf16, M x K), per CDNA5 ISA 16-bit A layout:
// lane l: m = l&15, group g = l>>4; VGPR v half h: K = (v>=4?16:0) + g*8 + (v&3)*2 + h
__device__ __forceinline__ v16bf load_fragA(const ushort_t* A, int rowStride, int m0, int k0, int lane) {
  const int m  = m0 + (lane & 15);
  const int lg = lane >> 4;
  const ushort_t* row = A + (size_t)m * rowStride + k0;
  FragU r;
#pragma unroll
  for (int v = 0; v < 8; ++v) {
    const int kp = ((v & 4) << 2) + (lg << 3) + ((v & 3) << 1); // even -> dword aligned pair
    r.u[v] = *(const uint_t*)(row + kp);
  }
  return r.v;
}

// B-fragment (32x16 bf16, K x N) from pre-packed buffer: 512 halves per fragment,
// lane-contiguous 32B -> two b128 loads.
__device__ __forceinline__ v16bf load_fragB(const ushort_t* P, int frag, int lane) {
  const ushort_t* p = P + (size_t)frag * 512 + lane * 16;
  FragU r;
  r.q[0] = *(const uint4*)(p);
  r.q[1] = *(const uint4*)(p + 8);
  return r.v;
}

#define WMMA_BF16(acc, a, b) \
  (acc) = __builtin_amdgcn_wmma_f32_16x16x32_bf16(false, (a), false, (b), (short)0, (acc), false, false)

// ---------------------------------------------------------------------------
// Pack a row-major f32 [K,N] weight into bf16 WMMA B-fragments.
// fragment f = kt*(N/16)+nt ; within fragment: lane*16 + j, where
// k = kt*32 + (j>>3)*16 + (lane>>4)*8 + (j&7) , n = nt*16 + (lane&15)
__global__ __launch_bounds__(256) void pack_b(const float* __restrict__ src,
                                              ushort_t* __restrict__ dst,
                                              int N, int total) {
  const int idx = blockIdx.x * 256 + threadIdx.x;
  if (idx >= total) return;
  const int f    = idx >> 9;
  const int r    = idx & 511;
  const int lane = r >> 4;
  const int j    = r & 15;
  const int Ntiles = N >> 4;
  const int kt = f / Ntiles;
  const int nt = f - kt * Ntiles;
  const int k = kt * 32 + ((j & 8) << 1) + ((lane >> 4) << 3) + (j & 7);
  const int n = (nt << 4) + (lane & 15);
  dst[idx] = f2bf(src[(size_t)k * N + n]);
}

// Embedding gather -> bf16 x [B*S, E]
__global__ __launch_bounds__(256) void embed_gather(const int* __restrict__ tokens,
                                                    const float* __restrict__ emb,
                                                    ushort_t* __restrict__ xb) {
  const int row = blockIdx.x;                 // b*S + s
  const int tok = tokens[row];
  const float* srcp = emb + (size_t)tok * EE;
  ushort_t*    dstp = xb  + (size_t)row * EE;
  for (int c = threadIdx.x; c < EE; c += 256) dstp[c] = f2bf(srcp[c]);
}

__global__ __launch_bounds__(256) void zero_u32(uint_t* p, int n) {
  const int i = blockIdx.x * 256 + threadIdx.x;
  if (i < n) p[i] = 0;
}

// ---------------------------------------------------------------------------
// Layer-1 bidirectional LSTM, one timestep, both directions in one launch.
// z = x_t @ W + h_prev @ U + b ; 256 waves: d(1) x mt(2b) x jt(5b)
__global__ __launch_bounds__(256) void lstm1_step(
    const ushort_t* __restrict__ xb,
    const ushort_t* __restrict__ pWf, const ushort_t* __restrict__ pUf, const float* __restrict__ bf_,
    const ushort_t* __restrict__ pWb, const ushort_t* __restrict__ pUb, const float* __restrict__ bb_,
    const ushort_t* __restrict__ hprev, ushort_t* __restrict__ hnext,
    float* __restrict__ cbuf, ushort_t* __restrict__ bi, int t) {
  const int lane = threadIdx.x & 31;
  const int w    = blockIdx.x * 8 + (threadIdx.x >> 5);
  const int d    = w >> 7;
  const int rem  = w & 127;
  const int m0   = (rem >> 5) << 4;      // batch tile base
  const int jt   = rem & 31;             // hidden 16-col tile
  const int j0   = jt << 4;
  const int t_eff = d ? (SS - 1 - t) : t;
  const ushort_t* pW  = d ? pWb : pWf;
  const ushort_t* pU  = d ? pUb : pUf;
  const float*    bia = d ? bb_ : bf_;
  const ushort_t* hp  = hprev + d * (BB * EE);

  v8f zi = {0,0,0,0,0,0,0,0}, zf = zi, zg = zi, zo = zi;

  // input contribution: A rows are batches at time t_eff
  const ushort_t* xbase = xb + (size_t)t_eff * EE;
#pragma unroll 2
  for (int kt = 0; kt < EE / 32; ++kt) {
    const int k0 = kt * 32;
    const v16bf a = load_fragA(xbase, SS * EE, m0, k0, lane);
    const int fb = kt * 128 + jt;        // Ntiles = 2048/16 = 128, gate stride 32
    if (kt + 1 < EE / 32) __builtin_prefetch(pW + (size_t)(fb + 128) * 512, 0, 1);
    WMMA_BF16(zi, a, load_fragB(pW, fb,      lane));
    WMMA_BF16(zf, a, load_fragB(pW, fb + 32, lane));
    WMMA_BF16(zg, a, load_fragB(pW, fb + 64, lane));
    WMMA_BF16(zo, a, load_fragB(pW, fb + 96, lane));
  }
  // recurrent contribution
#pragma unroll 2
  for (int kt = 0; kt < EE / 32; ++kt) {
    const int k0 = kt * 32;
    const v16bf a = load_fragA(hp, EE, m0, k0, lane);
    const int fb = kt * 128 + jt;
    if (kt + 1 < EE / 32) __builtin_prefetch(pU + (size_t)(fb + 128) * 512, 0, 1);
    WMMA_BF16(zi, a, load_fragB(pU, fb,      lane));
    WMMA_BF16(zf, a, load_fragB(pU, fb + 32, lane));
    WMMA_BF16(zg, a, load_fragB(pU, fb + 64, lane));
    WMMA_BF16(zo, a, load_fragB(pU, fb + 96, lane));
  }

  // gate math directly on accumulator layout: VGPR r, lane l -> M=r+8*(l>>4), N=l&15
  const int lg = lane >> 4;
  const int j  = j0 + (lane & 15);
#pragma unroll
  for (int r = 0; r < 8; ++r) {
    const int b  = m0 + r + (lg << 3);
    const float vi = zi[r] + bia[j];
    const float vf = zf[r] + bia[EE + j];
    const float vg = zg[r] + bia[2 * EE + j];
    const float vo = zo[r] + bia[3 * EE + j];
    const int cidx = d * (BB * EE) + b * EE + j;
    const float c_old = cbuf[cidx];
    const float cn = sigm(vf) * c_old + sigm(vi) * tanhf(vg);
    const float h  = sigm(vo) * tanhf(cn);
    cbuf[cidx] = cn;
    hnext[d * (BB * EE) + b * EE + j] = f2bf(h);
    bi[((size_t)b * SS + t_eff) * HH2 + d * EE + j] = f2bf(h);
  }
}

// ---------------------------------------------------------------------------
// Layer-2 LSTM (forward only), one timestep. z = bi_t @ W2 + h_prev @ U2 + b2
__global__ __launch_bounds__(256) void lstm2_step(
    const ushort_t* __restrict__ bi,
    const ushort_t* __restrict__ pW2, const ushort_t* __restrict__ pU2, const float* __restrict__ b2,
    const ushort_t* __restrict__ hprev, ushort_t* __restrict__ hnext,
    float* __restrict__ cbuf, float* __restrict__ hlast, int t) {
  const int lane = threadIdx.x & 31;
  const int w    = blockIdx.x * 8 + (threadIdx.x >> 5);
  const int m0   = (w >> 6) << 4;
  const int jt   = w & 63;
  const int j0   = jt << 4;

  v8f zi = {0,0,0,0,0,0,0,0}, zf = zi, zg = zi, zo = zi;

  const ushort_t* xbase = bi + (size_t)t * HH2;
#pragma unroll 2
  for (int kt = 0; kt < HH2 / 32; ++kt) {
    const int k0 = kt * 32;
    const v16bf a = load_fragA(xbase, SS * HH2, m0, k0, lane);
    const int fb = kt * 256 + jt;        // Ntiles = 4096/16 = 256, gate stride 64
    if (kt + 1 < HH2 / 32) __builtin_prefetch(pW2 + (size_t)(fb + 256) * 512, 0, 1);
    WMMA_BF16(zi, a, load_fragB(pW2, fb,       lane));
    WMMA_BF16(zf, a, load_fragB(pW2, fb + 64,  lane));
    WMMA_BF16(zg, a, load_fragB(pW2, fb + 128, lane));
    WMMA_BF16(zo, a, load_fragB(pW2, fb + 192, lane));
  }
#pragma unroll 2
  for (int kt = 0; kt < HH2 / 32; ++kt) {
    const int k0 = kt * 32;
    const v16bf a = load_fragA(hprev, HH2, m0, k0, lane);
    const int fb = kt * 256 + jt;
    if (kt + 1 < HH2 / 32) __builtin_prefetch(pU2 + (size_t)(fb + 256) * 512, 0, 1);
    WMMA_BF16(zi, a, load_fragB(pU2, fb,       lane));
    WMMA_BF16(zf, a, load_fragB(pU2, fb + 64,  lane));
    WMMA_BF16(zg, a, load_fragB(pU2, fb + 128, lane));
    WMMA_BF16(zo, a, load_fragB(pU2, fb + 192, lane));
  }

  const int lg = lane >> 4;
  const int j  = j0 + (lane & 15);
#pragma unroll
  for (int r = 0; r < 8; ++r) {
    const int b  = m0 + r + (lg << 3);
    const float vi = zi[r] + b2[j];
    const float vf = zf[r] + b2[HH2 + j];
    const float vg = zg[r] + b2[2 * HH2 + j];
    const float vo = zo[r] + b2[3 * HH2 + j];
    const int cidx = b * HH2 + j;
    const float c_old = cbuf[cidx];
    const float cn = sigm(vf) * c_old + sigm(vi) * tanhf(vg);
    const float h  = sigm(vo) * tanhf(cn);
    cbuf[cidx] = cn;
    hnext[cidx] = f2bf(h);
    hlast[cidx] = h;                     // last write (t=S-1) is the final h
  }
}

// ---------------------------------------------------------------------------
// Dense head: z = tanh(h_last @ Wd1 + bd1); out = softmax(z @ Wd2 + bd2)
__global__ __launch_bounds__(256) void head_kernel(
    const float* __restrict__ hlast, const float* __restrict__ Wd1, const float* __restrict__ bd1,
    const float* __restrict__ Wd2, const float* __restrict__ bd2, float* __restrict__ out) {
  __shared__ float sh[HH2];
  __shared__ float sz[HH2];
  __shared__ float pr[256 * LL];
  __shared__ float lgt[LL];
  const int b = blockIdx.x;
  const int tid = threadIdx.x;
  for (int k = tid; k < HH2; k += 256) sh[k] = hlast[b * HH2 + k];
  __syncthreads();
  for (int jj = tid; jj < HH2; jj += 256) {
    float acc = bd1[jj];
    for (int k = 0; k < HH2; ++k) acc = fmaf(sh[k], Wd1[(size_t)k * HH2 + jj], acc);
    sz[jj] = tanhf(acc);
  }
  __syncthreads();
  float part[LL];
#pragma unroll
  for (int l = 0; l < LL; ++l) part[l] = 0.f;
  for (int k = tid; k < HH2; k += 256) {
    const float z = sz[k];
#pragma unroll
    for (int l = 0; l < LL; ++l) part[l] = fmaf(z, Wd2[k * LL + l], part[l]);
  }
#pragma unroll
  for (int l = 0; l < LL; ++l) pr[tid * LL + l] = part[l];
  __syncthreads();
  if (tid < LL) {
    float acc = bd2[tid];
    for (int r = 0; r < 256; ++r) acc += pr[r * LL + tid];
    lgt[tid] = acc;
  }
  __syncthreads();
  if (tid == 0) {
    float mx = lgt[0];
    for (int l = 1; l < LL; ++l) mx = fmaxf(mx, lgt[l]);
    float e[LL], s = 0.f;
    for (int l = 0; l < LL; ++l) { e[l] = __expf(lgt[l] - mx); s += e[l]; }
    const float inv = 1.0f / s;
    for (int l = 0; l < LL; ++l) out[b * LL + l] = e[l] * inv;
  }
}

// ---------------------------------------------------------------------------
extern "C" void kernel_launch(void* const* d_in, const int* in_sizes, int n_in,
                              void* d_out, int out_size, void* d_ws, size_t ws_size,
                              hipStream_t stream) {
  (void)in_sizes; (void)n_in; (void)out_size; (void)ws_size;
  const int*   tokens = (const int*)  d_in[0];
  const float* emb    = (const float*)d_in[1];
  const float* W_f = (const float*)d_in[2];
  const float* U_f = (const float*)d_in[3];
  const float* b_f = (const float*)d_in[4];
  const float* W_b = (const float*)d_in[5];
  const float* U_b = (const float*)d_in[6];
  const float* b_b = (const float*)d_in[7];
  const float* W2  = (const float*)d_in[8];
  const float* U2  = (const float*)d_in[9];
  const float* b2  = (const float*)d_in[10];
  const float* Wd1 = (const float*)d_in[11];
  const float* bd1 = (const float*)d_in[12];
  const float* Wd2 = (const float*)d_in[13];
  const float* bd2 = (const float*)d_in[14];

  char* p = (char*)d_ws;
  auto take = [&](size_t bytes) { char* r = p; p += (bytes + 255) & ~(size_t)255; return r; };

  ushort_t* xb  = (ushort_t*)take((size_t)BB * SS * EE  * 2);  // 16 MB
  ushort_t* bi  = (ushort_t*)take((size_t)BB * SS * HH2 * 2);  // 32 MB
  ushort_t* pWf = (ushort_t*)take((size_t)EE  * 4 * EE  * 2);
  ushort_t* pUf = (ushort_t*)take((size_t)EE  * 4 * EE  * 2);
  ushort_t* pWb = (ushort_t*)take((size_t)EE  * 4 * EE  * 2);
  ushort_t* pUb = (ushort_t*)take((size_t)EE  * 4 * EE  * 2);
  ushort_t* pW2 = (ushort_t*)take((size_t)HH2 * 4 * HH2 * 2);
  ushort_t* pU2 = (ushort_t*)take((size_t)HH2 * 4 * HH2 * 2);
  // state region (zeroed contiguously each call)
  char* state0 = p;
  ushort_t* h1    = (ushort_t*)take((size_t)2 * 2 * BB * EE  * 2); // ping-pong x dir
  float*    c1    = (float*)   take((size_t)2 * BB * EE  * 4);
  ushort_t* h2    = (ushort_t*)take((size_t)2 * BB * HH2 * 2);
  float*    c2    = (float*)   take((size_t)BB * HH2 * 4);
  float*    hlast = (float*)   take((size_t)BB * HH2 * 4);
  const int state_u32 = (int)(((char*)p - state0) / 4);

  // 1) pack weights to bf16 WMMA B-fragment order (cheap vs 23.3 TB/s; L2 resident)
  pack_b<<<(EE * 4 * EE) / 256, 256, 0, stream>>>(W_f, pWf, 4 * EE, EE * 4 * EE);
  pack_b<<<(EE * 4 * EE) / 256, 256, 0, stream>>>(U_f, pUf, 4 * EE, EE * 4 * EE);
  pack_b<<<(EE * 4 * EE) / 256, 256, 0, stream>>>(W_b, pWb, 4 * EE, EE * 4 * EE);
  pack_b<<<(EE * 4 * EE) / 256, 256, 0, stream>>>(U_b, pUb, 4 * EE, EE * 4 * EE);
  pack_b<<<(HH2 * 4 * HH2) / 256, 256, 0, stream>>>(W2, pW2, 4 * HH2, HH2 * 4 * HH2);
  pack_b<<<(HH2 * 4 * HH2) / 256, 256, 0, stream>>>(U2, pU2, 4 * HH2, HH2 * 4 * HH2);

  // 2) embedding gather -> bf16
  embed_gather<<<BB * SS, 256, 0, stream>>>(tokens, emb, xb);

  // 3) zero recurrent state
  zero_u32<<<(state_u32 + 255) / 256, 256, 0, stream>>>((uint_t*)state0, state_u32);

  // 4) layer-1 bidirectional scan (fwd + bwd fused per launch)
  for (int t = 0; t < SS; ++t) {
    const ushort_t* hp = h1 + (size_t)(t & 1) * (2 * BB * EE);
    ushort_t*       hn = h1 + (size_t)((t + 1) & 1) * (2 * BB * EE);
    lstm1_step<<<32, 256, 0, stream>>>(xb, pWf, pUf, b_f, pWb, pUb, b_b, hp, hn, c1, bi, t);
  }

  // 5) layer-2 scan
  for (int t = 0; t < SS; ++t) {
    const ushort_t* hp = h2 + (size_t)(t & 1) * (BB * HH2);
    ushort_t*       hn = h2 + (size_t)((t + 1) & 1) * (BB * HH2);
    lstm2_step<<<32, 256, 0, stream>>>(bi, pW2, pU2, b2, hp, hn, c2, hlast, t);
  }

  // 6) dense head + softmax
  head_kernel<<<BB, 256, 0, stream>>>(hlast, Wd1, bd1, Wd2, bd2, (float*)d_out);
}
